// TemporalAttention_20057497272327
// MI455X (gfx1250) — matrix operand order repaired and checked by reading, not compiled, not run on hardware
//
#include <hip/hip_runtime.h>

typedef __bf16 bf16;
typedef __attribute__((ext_vector_type(16))) __bf16 v16bf;
typedef __attribute__((ext_vector_type(8)))  __bf16 v8bf;
typedef __attribute__((ext_vector_type(4)))  __bf16 v4bf;
typedef __attribute__((ext_vector_type(8)))  float  v8f;

#define WIN   64
#define DIM   512
#define HEADS 8
#define HD    64
#define STR_T 72    // bf16 tile stride (multiple of 8 -> 16B aligned rows)
#define STR_S 68    // f32 score stride
#define STR_Y 528   // bf16 Y stride (multiple of 8)

struct __align__(16) SMem {
  bf16  A  [WIN * STR_T];   // staged X tile (64 x 64)
  bf16  Wt0[WIN * STR_T];   // staged W^T tiles (64 n x 64 k)
  bf16  Wt1[WIN * STR_T];
  bf16  Wt2[WIN * STR_T];
  bf16  Q  [WIN * STR_T];   // Q head (64 q x 64 d)
  bf16  K  [WIN * STR_T];   // K head (64 key x 64 d)  == B^T for Q@K^T
  bf16  Vt [WIN * STR_T];   // V head transposed (64 d x 64 key) == B^T for P@V
  bf16  P  [WIN * STR_T];   // probs bf16 (64 q x 64 key)
  float S  [WIN * STR_S];   // raw scores f32
  bf16  Y  [WIN * STR_Y];   // attention output, heads concatenated (64 x 512)
};

// ---- WMMA fragment helpers (CDNA5 16x16x32 bf16 layouts) -------------------

__device__ __forceinline__ v16bf frag_A(const bf16* base, int stride, int lane) {
  int row = lane & 15;
  int kb  = (lane >> 4) << 3;            // 0 or 8
  const bf16* p = base + row * stride + kb;
  v8bf lo = *(const v8bf*)(p);
  v8bf hi = *(const v8bf*)(p + 16);
  return __builtin_shufflevector(lo, hi, 0,1,2,3,4,5,6,7,8,9,10,11,12,13,14,15);
}

__device__ __forceinline__ v16bf frag_Bt(const bf16* base, int stride, int lane) {
  int n  = lane & 15;
  int ko = (lane >> 4) << 3;             // 0 or 8
  const bf16* p = base + n * stride + ko;
  v8bf lo = *(const v8bf*)(p);
  v8bf hi = *(const v8bf*)(p + 16);
  return __builtin_shufflevector(lo, hi, 0,1,2,3,4,5,6,7,8,9,10,11,12,13,14,15);
}

__device__ __forceinline__ v8f wmma_bf16(v16bf a, v16bf b, v8f c) {
  return __builtin_amdgcn_wmma_f32_16x16x32_bf16(false, a, false, b, (short)0, c,
                                                 false, false);
}

// C/D 16x16 f32 layout: VGPR r, lanes 0-15 -> M=r, lanes 16-31 -> M=8+r, N=lane&15.
__device__ __forceinline__ void store_D_bf16(bf16* base, int stride, int lane,
                                             const v8f& d) {
  int n  = lane & 15;
  int m0 = (lane >> 4) << 3;
#pragma unroll
  for (int r = 0; r < 8; ++r) base[(m0 + r) * stride + n] = (bf16)d[r];
}
__device__ __forceinline__ void store_D_bf16_T(bf16* base, int stride, int lane,
                                               const v8f& d) {
  int n  = lane & 15;
  int m0 = (lane >> 4) << 3;
#pragma unroll
  for (int r = 0; r < 8; ++r) base[n * stride + (m0 + r)] = (bf16)d[r];
}
__device__ __forceinline__ void store_D_f32(float* base, int stride, int lane,
                                            const v8f& d) {
  int n  = lane & 15;
  int m0 = (lane >> 4) << 3;
#pragma unroll
  for (int r = 0; r < 8; ++r) base[(m0 + r) * stride + n] = d[r];
}

// ---- register-pipelined staging (float4 global loads, fp32 -> bf16) --------

// X tile: thread t loads float4 f = t&15 of rows (t>>4) + {0,16,32,48}.
__device__ __forceinline__ void load_x_f4(const float* __restrict__ xg, int row0,
                                          int k0, int tid, float4 r[4]) {
  int f  = tid & 15;
  int mg = tid >> 4;
#pragma unroll
  for (int i = 0; i < 4; ++i)
    r[i] = *(const float4*)(xg + (size_t)(row0 + mg + i * 16) * DIM + k0 + f * 4);
}
__device__ __forceinline__ void store_x_lds(bf16* dst, int tid, const float4 r[4]) {
  int f  = tid & 15;
  int mg = tid >> 4;
#pragma unroll
  for (int i = 0; i < 4; ++i) {
    v4bf t;
    t[0] = (bf16)r[i].x; t[1] = (bf16)r[i].y;
    t[2] = (bf16)r[i].z; t[3] = (bf16)r[i].w;
    *(v4bf*)(dst + (mg + i * 16) * STR_T + f * 4) = t;   // ds_store_b64
  }
}

// W tile: thread t loads float4 f = t&15 of k-rows (t>>4) + {0,16,32,48};
// stored transposed into Wt[n][k].
__device__ __forceinline__ void load_w_f4(const float* __restrict__ W, int ldw,
                                          int cb, int k0, int tid, float4 r[4]) {
  int f  = tid & 15;
  int kg = tid >> 4;
#pragma unroll
  for (int i = 0; i < 4; ++i)
    r[i] = *(const float4*)(W + (size_t)(k0 + kg + i * 16) * ldw + cb + f * 4);
}
__device__ __forceinline__ void store_wt_lds(bf16* dst, int tid, const float4 r[4]) {
  int f  = tid & 15;
  int kg = tid >> 4;
#pragma unroll
  for (int i = 0; i < 4; ++i) {
    int kk = kg + i * 16;
    dst[(f * 4 + 0) * STR_T + kk] = (bf16)r[i].x;
    dst[(f * 4 + 1) * STR_T + kk] = (bf16)r[i].y;
    dst[(f * 4 + 2) * STR_T + kk] = (bf16)r[i].z;
    dst[(f * 4 + 3) * STR_T + kk] = (bf16)r[i].w;
  }
}

// ---- kernel ----------------------------------------------------------------

__global__ __launch_bounds__(256) void win_attn_kernel(
    const float* __restrict__ x, const float* __restrict__ Wqkv,
    const float* __restrict__ Wproj, const float* __restrict__ bproj,
    float* __restrict__ out) {
  __shared__ SMem sm;
  const int tid  = threadIdx.x;
  const int lane = tid & 31;
  const int wv   = tid >> 5;         // 8 waves
  const int row0 = blockIdx.x * WIN;

  // two 16x16 tiles per wave; tI0 = 2w and 2w+1 share nt (shared B fragment).
  const int mt0 = (wv * 2) & 3;
  const int mt1 = mt0 + 1;
  const int nt  = (wv * 2) >> 2;

  bf16* const wt[3] = {sm.Wt0, sm.Wt1, sm.Wt2};

  // prime the staging pipeline: head 0, k-tile 0
  float4 ra[4], rw[3][4];
  load_x_f4(x, row0, 0, tid, ra);
#pragma unroll
  for (int q = 0; q < 3; ++q) load_w_f4(Wqkv, 3 * DIM, q * DIM, 0, tid, rw[q]);

  for (int h = 0; h < HEADS; ++h) {
    // ---- fused Q,K,V projections (register-pipelined staging) ----
    {
      v8f acc[3][2];
#pragma unroll
      for (int q = 0; q < 3; ++q) { acc[q][0] = {}; acc[q][1] = {}; }

      for (int k0 = 0; k0 < DIM; k0 += 64) {
        __syncthreads();
        store_x_lds(sm.A, tid, ra);
#pragma unroll
        for (int q = 0; q < 3; ++q) store_wt_lds(wt[q], tid, rw[q]);
        __syncthreads();

        // issue next tile's global loads; they overlap the WMMAs below
        {
          const bool last = (k0 + 64 >= DIM);
          const int  kn   = last ? 0 : k0 + 64;
          const int  hn   = last ? h + 1 : h;
          if (hn < HEADS) {
            load_x_f4(x, row0, kn, tid, ra);
#pragma unroll
            for (int q = 0; q < 3; ++q)
              load_w_f4(Wqkv, 3 * DIM, q * DIM + hn * HD, kn, tid, rw[q]);
          }
        }

#pragma unroll
        for (int ks = 0; ks < 2; ++ks) {
          v16bf a0 = frag_A(sm.A + mt0 * 16 * STR_T + ks * 32, STR_T, lane);
          v16bf a1 = frag_A(sm.A + mt1 * 16 * STR_T + ks * 32, STR_T, lane);
#pragma unroll
          for (int q = 0; q < 3; ++q) {
            v16bf b = frag_Bt(wt[q] + nt * 16 * STR_T + ks * 32, STR_T, lane);
            acc[q][0] = wmma_bf16(a0, b, acc[q][0]);
            acc[q][1] = wmma_bf16(a1, b, acc[q][1]);
          }
        }
      }
      store_D_bf16(sm.Q + mt0 * 16 * STR_T + nt * 16, STR_T, lane, acc[0][0]);
      store_D_bf16(sm.Q + mt1 * 16 * STR_T + nt * 16, STR_T, lane, acc[0][1]);
      store_D_bf16(sm.K + mt0 * 16 * STR_T + nt * 16, STR_T, lane, acc[1][0]);
      store_D_bf16(sm.K + mt1 * 16 * STR_T + nt * 16, STR_T, lane, acc[1][1]);
      // V stored transposed so P@V's B-fragments are contiguous
      store_D_bf16_T(sm.Vt + nt * 16 * STR_T + mt0 * 16, STR_T, lane, acc[2][0]);
      store_D_bf16_T(sm.Vt + nt * 16 * STR_T + mt1 * 16, STR_T, lane, acc[2][1]);
    }
    __syncthreads();

    // ---- S = Q @ K^T (64x64, d=64 => two 32-wide WMMA steps) ----
    {
      v8f acc0 = {}, acc1 = {};
#pragma unroll
      for (int ks = 0; ks < 2; ++ks) {
        v16bf b  = frag_Bt(sm.K + nt * 16 * STR_T + ks * 32, STR_T, lane);
        v16bf a0 = frag_A (sm.Q + mt0 * 16 * STR_T + ks * 32, STR_T, lane);
        v16bf a1 = frag_A (sm.Q + mt1 * 16 * STR_T + ks * 32, STR_T, lane);
        acc0 = wmma_bf16(a0, b, acc0);
        acc1 = wmma_bf16(a1, b, acc1);
      }
      store_D_f32(sm.S + mt0 * 16 * STR_S + nt * 16, STR_S, lane, acc0);
      store_D_f32(sm.S + mt1 * 16 * STR_S + nt * 16, STR_S, lane, acc1);
    }
    __syncthreads();

    // ---- causal mask + softmax, then the reference's second softmax over
    //      the probability row (masked entries participate as exact zeros) ----
    if (tid < WIN) {
      const int   r     = tid;
      const float scale = 0.125f;           // hd^-0.5, hd = 64
      float* srow = sm.S + r * STR_S;
      float mx = -3.4e38f;
      for (int j = 0; j <= r; ++j) mx = fmaxf(mx, srow[j] * scale);
      float sum = 0.f;
      for (int j = 0; j <= r; ++j) {
        float e = __expf(srow[j] * scale - mx);
        srow[j] = e;
        sum += e;
      }
      float inv = 1.f / sum;
      float mx2 = 0.f;                       // zeros are part of the row
      for (int j = 0; j <= r; ++j) mx2 = fmaxf(mx2, srow[j] * inv);
      float sum2 = 0.f;
      for (int j = 0; j < WIN; ++j) {
        float v = (j <= r) ? srow[j] * inv : 0.f;
        float e = __expf(v - mx2);
        srow[j] = e;
        sum2 += e;
      }
      float inv2 = 1.f / sum2;
      bf16* prow = sm.P + r * STR_T;
      for (int j = 0; j < WIN; ++j) prow[j] = (bf16)(srow[j] * inv2);
    }
    __syncthreads();

    // ---- O = P @ V, written into Y at this head's column block ----
    {
      v8f acc0 = {}, acc1 = {};
#pragma unroll
      for (int ks = 0; ks < 2; ++ks) {
        v16bf b  = frag_Bt(sm.Vt + nt * 16 * STR_T + ks * 32, STR_T, lane);
        v16bf a0 = frag_A (sm.P  + mt0 * 16 * STR_T + ks * 32, STR_T, lane);
        v16bf a1 = frag_A (sm.P  + mt1 * 16 * STR_T + ks * 32, STR_T, lane);
        acc0 = wmma_bf16(a0, b, acc0);
        acc1 = wmma_bf16(a1, b, acc1);
      }
      store_D_bf16(sm.Y + mt0 * 16 * STR_Y + h * HD + nt * 16, STR_Y, lane, acc0);
      store_D_bf16(sm.Y + mt1 * 16 * STR_Y + h * HD + nt * 16, STR_Y, lane, acc1);
    }
    __syncthreads();
  }

  // ---- out = Y @ Wproj + bproj, two 64-col chunks per pass (A reuse),
  //      register-pipelined W staging across chunk groups ----
  float4 rp[2][4];
  load_w_f4(Wproj, DIM, 0,  0, tid, rp[0]);
  load_w_f4(Wproj, DIM, 64, 0, tid, rp[1]);

  for (int cbg = 0; cbg < DIM; cbg += 128) {
    v8f acc[2][2];
#pragma unroll
    for (int c = 0; c < 2; ++c) { acc[c][0] = {}; acc[c][1] = {}; }

    for (int k0 = 0; k0 < DIM; k0 += 64) {
      __syncthreads();
      store_wt_lds(sm.Wt0, tid, rp[0]);
      store_wt_lds(sm.Wt1, tid, rp[1]);
      __syncthreads();

      {
        const bool last = (k0 + 64 >= DIM);
        const int  kn   = last ? 0 : k0 + 64;
        const int  cbn  = last ? cbg + 128 : cbg;
        if (cbn < DIM) {
          load_w_f4(Wproj, DIM, cbn,      kn, tid, rp[0]);
          load_w_f4(Wproj, DIM, cbn + 64, kn, tid, rp[1]);
        }
      }

#pragma unroll
      for (int ks = 0; ks < 2; ++ks) {
        v16bf a0 = frag_A(sm.Y + mt0 * 16 * STR_Y + k0 + ks * 32, STR_Y, lane);
        v16bf a1 = frag_A(sm.Y + mt1 * 16 * STR_Y + k0 + ks * 32, STR_Y, lane);
#pragma unroll
        for (int c = 0; c < 2; ++c) {
          v16bf b = frag_Bt(wt[c] + nt * 16 * STR_T + ks * 32, STR_T, lane);
          acc[c][0] = wmma_bf16(a0, b, acc[c][0]);
          acc[c][1] = wmma_bf16(a1, b, acc[c][1]);
        }
      }
    }
    const int n  = lane & 15;
    const int m0 = (lane >> 4) << 3;
#pragma unroll
    for (int c = 0; c < 2; ++c) {
      const int cb = cbg + c * 64;
      float bias = bproj[cb + nt * 16 + n];
#pragma unroll
      for (int r = 0; r < 8; ++r) {
        out[(size_t)(row0 + mt0 * 16 + m0 + r) * DIM + cb + nt * 16 + n] =
            acc[c][0][r] + bias;
        out[(size_t)(row0 + mt1 * 16 + m0 + r) * DIM + cb + nt * 16 + n] =
            acc[c][1][r] + bias;
      }
    }
  }
}

extern "C" void kernel_launch(void* const* d_in, const int* in_sizes, int n_in,
                              void* d_out, int out_size, void* d_ws, size_t ws_size,
                              hipStream_t stream) {
  const float* x     = (const float*)d_in[0];
  const float* Wqkv  = (const float*)d_in[1];
  const float* Wproj = (const float*)d_in[2];
  const float* bproj = (const float*)d_in[3];
  float*       out   = (float*)d_out;

  const int rows   = in_sizes[0] / DIM;   // 16 * 4096 = 65536
  const int blocks = rows / WIN;          // 1024 windows
  win_attn_kernel<<<blocks, 256, 0, stream>>>(x, Wqkv, Wproj, bproj, out);
}
